// GaranAttention_18038862643966
// MI455X (gfx1250) — matrix operand order repaired and checked by reading, not compiled, not run on hardware
//
#include <hip/hip_runtime.h>
#include <stdint.h>

// ---- CDNA5 WMMA types (probe-confirmed signatures) ----
typedef __attribute__((ext_vector_type(16))) __bf16 v16bf;
typedef __attribute__((ext_vector_type(8)))  float  v8f;

struct alignas(16) U128 { unsigned int x[4]; };
union V16 { v16bf v; U128 q[2]; unsigned short u[16]; };

__device__ __forceinline__ unsigned short f2bf(float f) {
    union { float f; uint32_t u; } x; x.f = f;
    uint32_t r = x.u + 0x7FFFu + ((x.u >> 16) & 1u);   // round-to-nearest-even
    return (unsigned short)(r >> 16);
}

// B=32, C=512, L=1024(HxW=32x32), nh=8, dh=64, temp=8

// ---------------------------------------------------------------------------
// K1: qp = q@Wq^T + bq ; fold qp into Wkc/Wkd rows -> aQK[b,16,512] (bf16)
//     rows 0..7 = collect heads, rows 8..15 = diffuse heads; qb = qp . bias
// ---------------------------------------------------------------------------
__global__ __launch_bounds__(256) void k_prep(
    const float* __restrict__ q, const float* __restrict__ Wq, const float* __restrict__ bq,
    const float* __restrict__ Wkc, const float* __restrict__ bkc,
    const float* __restrict__ Wkd, const float* __restrict__ bkd,
    unsigned short* __restrict__ aQK, float* __restrict__ qb)
{
    __shared__ float qrow[512];
    __shared__ float qp[512];
    const int b = blockIdx.x, tid = threadIdx.x;
    for (int i = tid; i < 512; i += 256) qrow[i] = q[b * 512 + i];
    __syncthreads();
    for (int o = tid; o < 512; o += 256) {
        float acc = bq[o];
        const float* wr = Wq + (size_t)o * 512;
        for (int c = 0; c < 512; ++c) acc += qrow[c] * wr[c];
        qp[o] = acc;
    }
    __syncthreads();
    for (int idx = tid; idx < 16 * 512; idx += 256) {
        const int m = idx >> 9, c = idx & 511, h = m & 7;
        const float* Wk = (m < 8) ? Wkc : Wkd;
        const float* qph = qp + h * 64;
        const float* wcol = Wk + (size_t)(h * 64) * 512 + c;
        float acc = 0.f;
        for (int d = 0; d < 64; ++d) acc += qph[d] * wcol[(size_t)d * 512];
        aQK[(b * 16 + m) * 512 + c] = f2bf(acc);
    }
    if (tid < 16) {
        const int m = tid, h = m & 7;
        const float* bias = (m < 8) ? bkc : bkd;
        float acc = 0.f;
        for (int d = 0; d < 64; ++d) acc += qp[h * 64 + d] * bias[h * 64 + d];
        qb[b * 16 + m] = acc;
    }
}

// ---------------------------------------------------------------------------
// K2: WMMA GEMM  D[16x16] += A(aQK 16x512) x B(v[b] 512x1024 tile)
//     rows 0..7 -> col_logit, rows 8..15 -> dif = sigmoid(logit)
// ---------------------------------------------------------------------------
__global__ __launch_bounds__(32) void k_logits(
    const float* __restrict__ v, const unsigned short* __restrict__ aQK,
    const float* __restrict__ qb, float* __restrict__ colLogit, float* __restrict__ dif)
{
    const int b = blockIdx.y;
    const int l0 = blockIdx.x * 16;
    const int lane = threadIdx.x;
    const int lh = lane >> 4, ll = lane & 15;
    v8f acc = {0.f, 0.f, 0.f, 0.f, 0.f, 0.f, 0.f, 0.f};
    const unsigned short* arow = aQK + (size_t)(b * 16 + ll) * 512 + lh * 8;
    const int crow = ll + 16 * lh;                 // unused; lane holds B column N = l0+ll
    for (int k0 = 0; k0 < 512; k0 += 32) {
        V16 A;                                     // A: K = k0 + (e>>3)*16 + lh*8 + (e&7)
        A.q[0] = *(const U128*)(arow + k0);
        A.q[1] = *(const U128*)(arow + k0 + 16);
        V16 Bm;                                    // B: N = l0+ll, K = k0 + (e>>3)*16 + lh*8 + (e&7)
        const float* bp = v + ((size_t)(b * 512 + k0 + lh * 8)) * 1024 + l0 + ll;
#pragma unroll
        for (int e = 0; e < 16; ++e)
            Bm.u[e] = f2bf(bp[(size_t)((e >> 3) * 16 + (e & 7)) * 1024]);
        acc = __builtin_amdgcn_wmma_f32_16x16x32_bf16(false, A.v, false, Bm.v,
                                                      (short)0, acc, false, false);
    }
    const int l = l0 + ll;                         // D: M = r + 8*lh, N = ll
    if (lh == 0) {
#pragma unroll
        for (int r = 0; r < 8; ++r) {
            float lg = (acc[r] + qb[b * 16 + r]) * 0.125f;
            colLogit[((size_t)(b * 8 + r)) * 1024 + l] = lg;
        }
    } else {
#pragma unroll
        for (int r = 0; r < 8; ++r) {
            float lg = (acc[r] + qb[b * 16 + 8 + r]) * 0.125f;
            dif[((size_t)(b * 8 + r)) * 1024 + l] = 1.0f / (1.0f + __expf(-lg));
        }
    }
}

// ---------------------------------------------------------------------------
// K3: softmax over L per (b,n) -> aAT rows 0..7 (bf16), rows 8..15 zeroed
// ---------------------------------------------------------------------------
__global__ __launch_bounds__(256) void k_softmax(
    const float* __restrict__ colLogit, unsigned short* __restrict__ aAT)
{
    __shared__ float red[256];
    const int row = blockIdx.x;                    // b*8 + n
    const int b = row >> 3, n = row & 7;
    const int tid = threadIdx.x;
    const float* src = colLogit + (size_t)row * 1024;
    float vals[4];
    float mx = -3.4e38f;
#pragma unroll
    for (int k = 0; k < 4; ++k) { vals[k] = src[tid + 256 * k]; mx = fmaxf(mx, vals[k]); }
    red[tid] = mx; __syncthreads();
    for (int s = 128; s > 0; s >>= 1) { if (tid < s) red[tid] = fmaxf(red[tid], red[tid + s]); __syncthreads(); }
    mx = red[0]; __syncthreads();
    float sum = 0.f;
#pragma unroll
    for (int k = 0; k < 4; ++k) { vals[k] = __expf(vals[k] - mx); sum += vals[k]; }
    red[tid] = sum; __syncthreads();
    for (int s = 128; s > 0; s >>= 1) { if (tid < s) red[tid] += red[tid + s]; __syncthreads(); }
    const float inv = 1.0f / red[0];
    unsigned short* dst  = aAT + (size_t)(b * 16 + n) * 1024;
    unsigned short* dstz = aAT + (size_t)(b * 16 + 8 + n) * 1024;
#pragma unroll
    for (int k = 0; k < 4; ++k) {
        dst[tid + 256 * k]  = f2bf(vals[k] * inv);
        dstz[tid + 256 * k] = 0;                   // zero-pad rows 8..15
    }
}

// ---------------------------------------------------------------------------
// K4: WMMA GEMM  t[b,8,512] = attn_col(16x1024) x v[b]^T (1024x512 tiles)
// ---------------------------------------------------------------------------
__global__ __launch_bounds__(32) void k_tmat(
    const float* __restrict__ v, const unsigned short* __restrict__ aAT,
    float* __restrict__ t)
{
    const int b = blockIdx.y;
    const int c0 = blockIdx.x * 16;
    const int lane = threadIdx.x;
    const int lh = lane >> 4, ll = lane & 15;
    v8f acc = {0.f, 0.f, 0.f, 0.f, 0.f, 0.f, 0.f, 0.f};
    const unsigned short* arow = aAT + (size_t)(b * 16 + ll) * 1024 + lh * 8;
    const float* vb = v + (size_t)b * 512 * 1024;
    for (int k0 = 0; k0 < 1024; k0 += 32) {
        V16 A;
        A.q[0] = *(const U128*)(arow + k0);
        A.q[1] = *(const U128*)(arow + k0 + 16);
        const int kl = k0 + lh * 8;                // K = kl + (e>>3)*16 + (e&7)
        V16 Bm;
#pragma unroll
        for (int e = 0; e < 16; ++e)               // N = channel c0+ll
            Bm.u[e] = f2bf(vb[(size_t)(c0 + ll) * 1024 + kl + (e >> 3) * 16 + (e & 7)]);
        acc = __builtin_amdgcn_wmma_f32_16x16x32_bf16(false, A.v, false, Bm.v,
                                                      (short)0, acc, false, false);
    }
    if (lh == 0) {                                  // M = r (0..7 valid), N = ll
#pragma unroll
        for (int r = 0; r < 8; ++r)
            t[((size_t)(b * 8 + r)) * 512 + c0 + ll] = acc[r];
    }
}

// ---------------------------------------------------------------------------
// K5: attn[b,o] = Wvs.t[b] + bvs ; woa[b,o,n] = Wo[o, n*64..]·attn[b,n*64..]
// ---------------------------------------------------------------------------
__global__ __launch_bounds__(256) void k_attn_woa(
    const float* __restrict__ t, const float* __restrict__ Wvs, const float* __restrict__ bvs,
    const float* __restrict__ Wo, float* __restrict__ woa)
{
    __shared__ float tl[8 * 512];
    __shared__ float attn[512];
    const int b = blockIdx.x, tid = threadIdx.x;
    for (int i = tid; i < 8 * 512; i += 256) tl[i] = t[(size_t)b * 8 * 512 + i];
    __syncthreads();
    for (int o = tid; o < 512; o += 256) {
        const int n = o >> 6;
        float acc = bvs[o];                         // bvs * sum(attn_col)=bvs
        const float* wr = Wvs + (size_t)o * 512;
        const float* tr = tl + n * 512;
        for (int c = 0; c < 512; ++c) acc += wr[c] * tr[c];
        attn[o] = acc;
    }
    __syncthreads();
    for (int idx = tid; idx < 512 * 8; idx += 256) {
        const int o = idx >> 3, n = idx & 7;
        const float* wr = Wo + (size_t)o * 512 + n * 64;
        const float* ar = attn + n * 64;
        float acc = 0.f;
#pragma unroll 8
        for (int d = 0; d < 64; ++d) acc += wr[d] * ar[d];
        woa[(size_t)b * 4096 + idx] = acc;
    }
}

// ---------------------------------------------------------------------------
// K6: attn_map[b,o,i,j] = sum_n dif[b,n, j*32+i] * woa[b,o,n] + bo[o]
//     (spatial transpose from the reference's permute(0,1,4,3,2))
// ---------------------------------------------------------------------------
__global__ __launch_bounds__(256) void k_amap(
    const float* __restrict__ dif, const float* __restrict__ woa,
    const float* __restrict__ bo, float* __restrict__ amap)
{
    __shared__ float dl[8 * 1024];
    __shared__ float wl[8][8];
    const int b = blockIdx.y;
    const int o0 = blockIdx.x * 8;
    const int tid = threadIdx.x;
    for (int i = tid; i < 8 * 1024; i += 256) dl[i] = dif[(size_t)b * 8192 + i];
    if (tid < 64) wl[tid >> 3][tid & 7] = woa[(size_t)b * 4096 + (size_t)(o0 + (tid >> 3)) * 8 + (tid & 7)];
    __syncthreads();
#pragma unroll
    for (int oo = 0; oo < 8; ++oo) {
        const int o = o0 + oo;
        const float bias = bo[o];
        for (int k = 0; k < 4; ++k) {
            const int s = tid + 256 * k;
            const int sT = ((s & 31) << 5) | (s >> 5);
            float acc = bias;
#pragma unroll
            for (int n = 0; n < 8; ++n) acc += dl[n * 1024 + sT] * wl[oo][n];
            amap[((size_t)(b * 512 + o)) * 1024 + s] = acc;
        }
    }
}

// ---------------------------------------------------------------------------
// K7: BatchNorm batch stats per channel (deterministic two-pass-free reduce)
// ---------------------------------------------------------------------------
__global__ __launch_bounds__(256) void k_bnstats(
    const float* __restrict__ amap, const float* __restrict__ gamma,
    const float* __restrict__ beta, float* __restrict__ scale, float* __restrict__ shift)
{
    __shared__ float s1[256], s2[256];
    const int o = blockIdx.x, tid = threadIdx.x;
    float a = 0.f, a2 = 0.f;
    for (int idx = tid; idx < 32 * 1024; idx += 256) {
        const int b = idx >> 10, s = idx & 1023;
        const float x = amap[((size_t)(b * 512 + o)) * 1024 + s];
        a += x; a2 += x * x;
    }
    s1[tid] = a; s2[tid] = a2; __syncthreads();
    for (int st = 128; st > 0; st >>= 1) {
        if (tid < st) { s1[tid] += s1[tid + st]; s2[tid] += s2[tid + st]; }
        __syncthreads();
    }
    if (tid == 0) {
        const float mean = s1[0] * (1.0f / 32768.0f);
        const float var  = s2[0] * (1.0f / 32768.0f) - mean * mean;
        const float sc   = gamma[o] * rsqrtf(var + 1e-5f);
        scale[o] = sc;
        shift[o] = beta[o] - mean * sc;
    }
}

// ---------------------------------------------------------------------------
// K8: m_attn = 3x3 conv (d_o -> 1, SAME) over attn_map, LDS-staged per plane
// ---------------------------------------------------------------------------
__global__ __launch_bounds__(1024) void k_mattn(
    const float* __restrict__ amap, const float* __restrict__ Wm,
    const float* __restrict__ bm, float* __restrict__ m_attn)
{
    __shared__ float plane[1024];
    const int b = blockIdx.x, tid = threadIdx.x;
    const int i = tid >> 5, j = tid & 31;
    float acc = bm[0];
    for (int o = 0; o < 512; ++o) {
        __syncthreads();
        plane[tid] = amap[((size_t)(b * 512 + o)) * 1024 + tid];
        __syncthreads();
        const float* w = Wm + o * 9;
#pragma unroll
        for (int di = 0; di < 3; ++di) {
            const int ii = i + di - 1;
            if (ii < 0 || ii > 31) continue;
#pragma unroll
            for (int dj = 0; dj < 3; ++dj) {
                const int jj = j + dj - 1;
                if (jj < 0 || jj > 31) continue;
                acc += plane[ii * 32 + jj] * w[di * 3 + dj];
            }
        }
    }
    m_attn[b * 1024 + tid] = acc;
}

// ---------------------------------------------------------------------------
// K9: out = LeakyReLU( BN(attn_map) + residual v )
// ---------------------------------------------------------------------------
__global__ __launch_bounds__(256) void k_final(
    const float* __restrict__ amap, const float* __restrict__ v,
    const float* __restrict__ scale, const float* __restrict__ shift,
    float* __restrict__ out)
{
    const size_t idx = (size_t)blockIdx.x * 256 + threadIdx.x;
    if (idx >= (size_t)32 * 512 * 1024) return;
    const int o = (int)((idx >> 10) & 511);
    const float x = amap[idx] * scale[o] + shift[o] + v[idx];
    out[idx] = x >= 0.f ? x : 0.1f * x;
}

// ---------------------------------------------------------------------------
extern "C" void kernel_launch(void* const* d_in, const int* in_sizes, int n_in,
                              void* d_out, int out_size, void* d_ws, size_t ws_size,
                              hipStream_t stream)
{
    const float* q     = (const float*)d_in[0];
    const float* v     = (const float*)d_in[1];
    const float* Wq    = (const float*)d_in[2];
    const float* bq    = (const float*)d_in[3];
    const float* Wkc   = (const float*)d_in[4];
    const float* bkc   = (const float*)d_in[5];
    const float* Wkd   = (const float*)d_in[6];
    const float* bkd   = (const float*)d_in[7];
    const float* Wvs   = (const float*)d_in[8];
    const float* bvs   = (const float*)d_in[9];
    const float* Wo    = (const float*)d_in[10];
    const float* bo    = (const float*)d_in[11];
    const float* Wm    = (const float*)d_in[12];
    const float* bm    = (const float*)d_in[13];
    const float* gamma = (const float*)d_in[14];
    const float* beta  = (const float*)d_in[15];

    float* out    = (float*)d_out;        // (32,512,32,32)
    float* m_attn = out + 16777216;       // (32,1024)
    float* amap   = m_attn + 32768;       // (32,512,32,32)

    char* w = (char*)d_ws;
    unsigned short* aQK = (unsigned short*)(w + 0);        // 32*16*512 bf16
    float* qb           = (float*)(w + 524288);            // 32*16
    float* colLogit     = (float*)(w + 526336);            // 32*8*1024
    float* dif          = (float*)(w + 1574912);           // 32*8*1024
    unsigned short* aAT = (unsigned short*)(w + 2623488);  // 32*16*1024 bf16
    float* t            = (float*)(w + 3672064);           // 32*8*512
    float* woa          = (float*)(w + 4196352);           // 32*512*8
    float* bnscale      = (float*)(w + 4720640);           // 512
    float* bnshift      = (float*)(w + 4722688);           // 512

    k_prep    <<<32,          256, 0, stream>>>(q, Wq, bq, Wkc, bkc, Wkd, bkd, aQK, qb);
    k_logits  <<<dim3(64,32),  32, 0, stream>>>(v, aQK, qb, colLogit, dif);
    k_softmax <<<256,         256, 0, stream>>>(colLogit, aAT);
    k_tmat    <<<dim3(32,32),  32, 0, stream>>>(v, aAT, t);
    k_attn_woa<<<32,          256, 0, stream>>>(t, Wvs, bvs, Wo, woa);
    k_amap    <<<dim3(64,32), 256, 0, stream>>>(dif, woa, bo, amap);
    k_bnstats <<<512,         256, 0, stream>>>(amap, gamma, beta, bnscale, bnshift);
    k_mattn   <<<32,         1024, 0, stream>>>(amap, Wm, bm, m_attn);
    k_final   <<<65536,       256, 0, stream>>>(amap, v, bnscale, bnshift, out);
}
